// TKAN_25262997636019
// MI455X (gfx1250) — matrix operand order repaired and checked by reading, not compiled, not run on hardware
//
#include <hip/hip_runtime.h>
#include <hip/hip_bf16.h>

typedef __attribute__((ext_vector_type(16))) _Float16 v16h;
typedef __attribute__((ext_vector_type(8)))  _Float16 v8h;
typedef __attribute__((ext_vector_type(8)))  float    v8f;

#define B_ 64
#define T_ 512
#define D_ 512
#define H_ 1024
#define K_ 3
#define CIN_ (D_ + H_)        // 1536 reduction length
#define NCHUNK_ (CIN_ / 32)   // 48 k-chunks of 32
#define XCHUNK_ (D_ / 32)     // 16 chunks sourced from x, rest from h
#define NT_ (H_ / 16)         // 64 N tiles
#define MT_ (B_ / 16)         // 4  M tiles
#define WPT_ 4                // waves per tile (split-K factor)
#define CPW_ (NCHUNK_ / WPT_) // 12 chunks per wave
#define KSTRIDE_ 1572864      // elements between weight branches (NT_*NCHUNK_*512)

// ---------------- preprocessing kernels ----------------

__global__ void cvt_x_f16_kernel(const float* __restrict__ x,
                                 _Float16* __restrict__ xf, int n) {
  for (int i = blockIdx.x * blockDim.x + threadIdx.x; i < n;
       i += gridDim.x * blockDim.x)
    xf[i] = (_Float16)x[i];
}

// Repack W[k][i][h] (f32, row-major) into the per-lane fragment order of
// V_WMMA_F32_16X16X32_F16's B operand (32x16 f16): lanes 0-15: N=lane,
// K=0..7,16..23 packed; lanes 16-31: N=lane-16, K=8..15,24..31.
// Layout: wf[(((k*NT + nt)*NCHUNK + chunk)*32 + lane)*16 + e]
__global__ void build_wfrag_kernel(const float* __restrict__ W,
                                   _Float16* __restrict__ wf, int n) {
  for (int i = blockIdx.x * blockDim.x + threadIdx.x; i < n;
       i += gridDim.x * blockDim.x) {
    int e     = i & 15;
    int lane  = (i >> 4) & 31;
    int chunk = (i >> 9) % NCHUNK_;
    int rest  = (i >> 9) / NCHUNK_;
    int nt    = rest % NT_;
    int k     = rest / NT_;
    int ncol  = nt * 16 + (lane & 15);
    int kk    = chunk * 32 + ((lane >> 4) << 3) + (e & 7) + ((e >> 3) << 4);
    wf[i] = (_Float16)W[((size_t)k * CIN_ + kk) * H_ + ncol];
  }
}

__global__ void init_h_kernel(const float* __restrict__ h0,
                              _Float16* __restrict__ hf, int n) {
  for (int i = blockIdx.x * blockDim.x + threadIdx.x; i < n;
       i += gridDim.x * blockDim.x)
    hf[i] = (_Float16)h0[i % H_];   // broadcast [1,H] over batch
}

// ---------------- per-timestep GEMM + ReLU-sum step ----------------

// WGP-scope prefetch (no scope modifier == SCOPE_CU): pulls into ALL cache
// levels, unlike the SE-scope prefetch __builtin_prefetch emits (GL2-only,
// useless here since W is already L2-resident).
__device__ __forceinline__ void prefetch_cu(const void* p) {
  asm volatile("global_prefetch_b8 %0, off" ::"v"(p));
}
__device__ __forceinline__ void prefetch_cu_b1(const void* p) {
  asm volatile("global_prefetch_b8 %0, off offset:3145728" ::"v"(p));
}
__device__ __forceinline__ void prefetch_cu_b2(const void* p) {
  asm volatile("global_prefetch_b8 %0, off offset:6291456" ::"v"(p));
}

// Compile-time-specialized chunk loop: NX chunks sourced from x, then NH
// from h. All selects/offsets are static after unroll -> pure
// base+immediate addressing, no per-iteration cndmask chains.
template <int NX, int NH>
__device__ __forceinline__ void kchunks(const _Float16* __restrict__ pax,
                                        const _Float16* __restrict__ pah,
                                        const _Float16* __restrict__ pb0,
                                        v8f& acc0, v8f& acc1, v8f& acc2) {
  constexpr int NC = NX + NH;
  if constexpr (NC > 1) {  // warm chunk 1 (chunk 0's loads issue immediately)
    const _Float16* pf = pb0 + 512;
    prefetch_cu(pf);
    prefetch_cu_b1(pf);
    prefetch_cu_b2(pf);
  }
#pragma unroll
  for (int i = 0; i < NC; ++i) {
    if (i + 2 < NC) {  // prefetch distance 2: ~6 WMMAs between warm and use
      const _Float16* pf = pb0 + (size_t)(i + 2) * 512;
      prefetch_cu(pf);
      prefetch_cu_b1(pf);
      prefetch_cu_b2(pf);
      const _Float16* pan =
          (i + 2 < NX) ? (pax + (i + 2) * 32) : (pah + (i + 2 - NX) * 32);
      prefetch_cu(pan);
    }
    const _Float16* pa = (i < NX) ? (pax + i * 32) : (pah + (i - NX) * 32);
    v16h a;
    *((v8h*)&a)     = *(const v8h*)pa;
    *((v8h*)&a + 1) = *(const v8h*)(pa + 16);
    v16h b0 = *(const v16h*)(pb0 + (size_t)i * 512);
    v16h b1 = *(const v16h*)(pb0 + KSTRIDE_ + (size_t)i * 512);
    v16h b2 = *(const v16h*)(pb0 + 2 * KSTRIDE_ + (size_t)i * 512);

    acc0 = __builtin_amdgcn_wmma_f32_16x16x32_f16(false, a, false, b0,
                                                  (short)0, acc0, false, false);
    acc1 = __builtin_amdgcn_wmma_f32_16x16x32_f16(false, a, false, b1,
                                                  (short)0, acc1, false, false);
    acc2 = __builtin_amdgcn_wmma_f32_16x16x32_f16(false, a, false, b2,
                                                  (short)0, acc2, false, false);
  }
}

__global__ __launch_bounds__(128) void tkan_step_kernel(
    const _Float16* __restrict__ xf,    // [B,T,D] f16
    const _Float16* __restrict__ wf,    // fragment-packed weights
    const float*    __restrict__ bias,  // [K,H] f32
    const _Float16* __restrict__ hcur,  // [B,H] f16 (state at t)
    _Float16*       __restrict__ hnext, // [B,H] f16 (state at t+1)
    float*          __restrict__ out,   // [B,T,H] f32
    int t) {
  // One workgroup (4 waves) per 16x16 output tile; split-K across waves.
  const int lane  = threadIdx.x & 31;
  const int wave  = threadIdx.x >> 5;          // 0..3 (k-segment)
  const int mtile = blockIdx.x >> 6;           // 0..3
  const int ntile = blockIdx.x & 63;           // 0..63
  const int cbase = wave * CPW_;

  // A addressing: lane L -> row M=L%16; half=L/16 selects K sub-ranges.
  const int mrow = lane & 15;
  const int half = lane >> 4;
  const int brow = mtile * 16 + mrow;

  const _Float16* xrow = xf + ((size_t)brow * T_ + t) * D_ + half * 8;
  const _Float16* hrow = hcur + (size_t)brow * H_ + half * 8;

  const _Float16* wfb =
      wf + (size_t)ntile * NCHUNK_ * 512 + lane * 16;  // branch-0 base

  // Hoist bias loads (needed only in the epilogue).
  const int ncol = ntile * 16 + (lane & 15);
  const float b0s = bias[0 * H_ + ncol];
  const float b1s = bias[1 * H_ + ncol];
  const float b2s = bias[2 * H_ + ncol];

  v8f acc0 = {}, acc1 = {}, acc2 = {};

  const _Float16* pax = xrow + cbase * 32;  // meaningful when cbase < 16
  const _Float16* pah =
      hrow + ((cbase >= XCHUNK_) ? (cbase - XCHUNK_) * 32 : 0);
  const _Float16* pb0 = wfb + (size_t)cbase * 512;

  if (wave == 0)
    kchunks<12, 0>(pax, pah, pb0, acc0, acc1, acc2);   // chunks 0-11: all x
  else if (wave == 1)
    kchunks<4, 8>(pax, pah, pb0, acc0, acc1, acc2);    // 12-15 x, 16-23 h
  else
    kchunks<0, 12>(pax, pah, pb0, acc0, acc1, acc2);   // all h

  // ---- split-K reduction through LDS ----
  // red[wave][branch][lane][r]: per-lane contiguous r -> vectorizable stores.
  __shared__ float red[WPT_][K_][32][8];
#pragma unroll
  for (int r = 0; r < 8; ++r) red[wave][0][lane][r] = acc0[r];
#pragma unroll
  for (int r = 0; r < 8; ++r) red[wave][1][lane][r] = acc1[r];
#pragma unroll
  for (int r = 0; r < 8; ++r) red[wave][2][lane][r] = acc2[r];

  __syncthreads();

  // Wave w reduces C rows r in {2w, 2w+1} across the 4 partials, then does
  // bias + ReLU + branch-sum and writes its 2 batch rows.
  const int r0 = 2 * wave;
  float p00 = 0.f, p01 = 0.f, p10 = 0.f, p11 = 0.f, p20 = 0.f, p21 = 0.f;
#pragma unroll
  for (int sw = 0; sw < WPT_; ++sw) {
    p00 += red[sw][0][lane][r0];
    p01 += red[sw][0][lane][r0 + 1];
    p10 += red[sw][1][lane][r0];
    p11 += red[sw][1][lane][r0 + 1];
    p20 += red[sw][2][lane][r0];
    p21 += red[sw][2][lane][r0 + 1];
  }

  const float hv0 = fmaxf(p00 + b0s, 0.f) + fmaxf(p10 + b1s, 0.f) +
                    fmaxf(p20 + b2s, 0.f);
  const float hv1 = fmaxf(p01 + b0s, 0.f) + fmaxf(p11 + b1s, 0.f) +
                    fmaxf(p21 + b2s, 0.f);

  // C/D layout: VGPR r, lanes 0-15 -> (M=r, N=lane); lanes 16-31 -> (M=r+8).
  const int bg0 = mtile * 16 + ((lane < 16) ? r0 : r0 + 8);
  out[((size_t)bg0 * T_ + t) * H_ + ncol]       = hv0;
  out[((size_t)(bg0 + 1) * T_ + t) * H_ + ncol] = hv1;
  hnext[(size_t)bg0 * H_ + ncol]       = (_Float16)hv0;
  hnext[(size_t)(bg0 + 1) * H_ + ncol] = (_Float16)hv1;
}

// ---------------- launcher ----------------

extern "C" void kernel_launch(void* const* d_in, const int* in_sizes, int n_in,
                              void* d_out, int out_size, void* d_ws,
                              size_t ws_size, hipStream_t stream) {
  (void)in_sizes; (void)n_in; (void)out_size; (void)ws_size;
  const float* x  = (const float*)d_in[0];  // [B,T,D]
  const float* W  = (const float*)d_in[1];  // [K,D+H,H]
  const float* b  = (const float*)d_in[2];  // [K,H]
  const float* h0 = (const float*)d_in[3];  // [1,H]

  char* ws = (char*)d_ws;
  _Float16* xf    = (_Float16*)ws;                           // 33,554,432 B
  _Float16* wfrag = (_Float16*)(ws + 33554432);              //  9,437,184 B
  _Float16* hbuf0 = (_Float16*)(ws + 33554432 + 9437184);    //    131,072 B
  _Float16* hbuf1 = hbuf0 + (size_t)B_ * H_;                 //    131,072 B

  const int nx = B_ * T_ * D_;
  cvt_x_f16_kernel<<<2048, 256, 0, stream>>>(x, xf, nx);

  const int nw = K_ * NT_ * NCHUNK_ * 32 * 16;
  build_wfrag_kernel<<<2048, 256, 0, stream>>>(W, wfrag, nw);

  init_h_kernel<<<256, 256, 0, stream>>>(h0, hbuf0, B_ * H_);

  float* out = (float*)d_out;
  for (int t = 0; t < T_; ++t) {
    const _Float16* hc = (t & 1) ? hbuf1 : hbuf0;
    _Float16*       hn = (t & 1) ? hbuf0 : hbuf1;
    tkan_step_kernel<<<MT_ * NT_, 128, 0, stream>>>(xf, wfrag, b, hc, hn, out,
                                                    t);
  }
}
